// Seq2seqModel_8100308320460
// MI455X (gfx1250) — matrix operand order repaired and checked by reading, not compiled, not run on hardware
//
#include <hip/hip_runtime.h>

#define HH 1024
#define EE 256
#define BB 64
#define SS 128
#define TT 64
#define AAV 2048
#define G4H 4096

typedef float v8f __attribute__((ext_vector_type(8)));
typedef __bf16 v16bf __attribute__((ext_vector_type(16)));

union Frag { v16bf v; uint4 q[2]; };

__device__ __forceinline__ unsigned short f2bf(float f) {
  unsigned int u = __float_as_uint(f);
  u += 0x7fffu + ((u >> 16) & 1u);
  return (unsigned short)(u >> 16);
}
__device__ __forceinline__ float sigm(float x) { return 1.0f / (1.0f + expf(-x)); }

// ---------------------------------------------------------------------------
// Pack a weight matrix into WMMA B-fragment tile order (bf16).
// Logical Bmat[k,n], k<K, n<N.  transp=1: src is [N,K] (used as x @ W.T).
// Tile (nt,kt): 512 halves, element (lane,h): k = kt*32 + h + 16*(lane>>4),
// n = nt*16 + (lane&15). Tile order: tile = nt*(K/32) + kt.
// ---------------------------------------------------------------------------
__global__ void pack_w(const float* __restrict__ src, unsigned short* __restrict__ dst,
                       int K, int N, int transp) {
  long i = (long)blockIdx.x * blockDim.x + threadIdx.x;
  long total = (long)K * N;
  if (i >= total) return;
  long tile = i >> 9;
  int within = (int)(i & 511);
  int lane = within >> 4, h = within & 15;
  int nK = K >> 5;
  int kt = (int)(tile % nK), nt = (int)(tile / nK);
  int k = kt * 32 + h + 16 * (lane >> 4);
  int n = nt * 16 + (lane & 15);
  float v = transp ? src[(long)n * K + k] : src[(long)k * N + n];
  dst[i] = f2bf(v);
}

// ---------------------------------------------------------------------------
// Generic bf16 WMMA GEMM:  C = A1 @ W1 (+ A2 @ W2) (+ bias), optional tanh.
// Each wave computes a 2x2 block of 16x16 tiles (32x32 of C): per K-step
// 2 A-frags + 2 B-frags (8x b128) feed 4 v_wmma -> 2 loads/WMMA.
// Requires M%32==0 && N%32==0 (true for all call sites).
// blockIdx.y = direction; per-direction element strides sW1/sW2/sA2/sBias/sCf.
// timeMode=1: A1 rows are time-indexed encoder activations [S,B,K1]:
//   fwd (dir 0) reads time t, bwd (dir 1) reads clamp(len[b]-1-t, 0, S-1).
// ---------------------------------------------------------------------------
__global__ void wmma_gemm(
    const unsigned short* __restrict__ A1, int K1, const unsigned short* __restrict__ W1,
    const unsigned short* __restrict__ A2, int K2, const unsigned short* __restrict__ W2,
    const float* __restrict__ bias,
    float* __restrict__ Cf, int ldc,
    unsigned short* __restrict__ Cb, int ldcb,
    int M, int N, int act,
    long sW1, long sW2, long sA2, int sBias, long sCf,
    const int* __restrict__ lens, int t, int timeMode)
{
  int dir  = blockIdx.y;
  int wave = threadIdx.x >> 5;
  int lane = threadIdx.x & 31;
  int Mt2 = M >> 5, Nt2 = N >> 5;
  int tile = blockIdx.x * 4 + wave;
  if (tile >= Mt2 * Nt2) return;
  int mt = (tile % Mt2) * 2;        // first of two 16-row tiles
  int nt = (tile / Mt2) * 2;        // first of two 16-col tiles
  int lrow = lane & 15, hi = lane >> 4;
  int m0 = mt * 16 + lrow;
  int m1 = m0 + 16;

  const unsigned short *arowA0, *arowA1;
  if (timeMode) {
    int len0 = lens[m0];
    int s0 = t;
    if (dir) { s0 = len0 - 1 - t; if (s0 < 0) s0 = 0; if (s0 > SS - 1) s0 = SS - 1; }
    arowA0 = A1 + ((long)(s0 * BB + m0)) * K1;
    int len1 = lens[m1];
    int s1 = t;
    if (dir) { s1 = len1 - 1 - t; if (s1 < 0) s1 = 0; if (s1 > SS - 1) s1 = SS - 1; }
    arowA1 = A1 + ((long)(s1 * BB + m1)) * K1;
  } else {
    arowA0 = A1 + (long)m0 * K1;
    arowA1 = A1 + (long)m1 * K1;
  }

  v8f acc00 = {0.f,0.f,0.f,0.f,0.f,0.f,0.f,0.f};
  v8f acc01 = acc00, acc10 = acc00, acc11 = acc00;

  int nK1 = K1 >> 5;
  const unsigned short* w1a = W1 + dir * sW1 + ((long)nt * nK1) * 512 + lane * 16;
  const unsigned short* w1b = w1a + (long)nK1 * 512;
  for (int kt = 0; kt < nK1; ++kt) {
    Frag a0, a1, b0, b1;
    const unsigned short* ap0 = arowA0 + kt * 32 + 16 * hi;
    const unsigned short* ap1 = arowA1 + kt * 32 + 16 * hi;
    a0.q[0] = *(const uint4*)(ap0);
    a0.q[1] = *(const uint4*)(ap0 + 8);
    a1.q[0] = *(const uint4*)(ap1);
    a1.q[1] = *(const uint4*)(ap1 + 8);
    b0.q[0] = *(const uint4*)(w1a);
    b0.q[1] = *(const uint4*)(w1a + 8);
    b1.q[0] = *(const uint4*)(w1b);
    b1.q[1] = *(const uint4*)(w1b + 8);
    w1a += 512; w1b += 512;
    acc00 = __builtin_amdgcn_wmma_f32_16x16x32_bf16(false, a0.v, false, b0.v, (short)0, acc00, false, false);
    acc01 = __builtin_amdgcn_wmma_f32_16x16x32_bf16(false, a0.v, false, b1.v, (short)0, acc01, false, false);
    acc10 = __builtin_amdgcn_wmma_f32_16x16x32_bf16(false, a1.v, false, b0.v, (short)0, acc10, false, false);
    acc11 = __builtin_amdgcn_wmma_f32_16x16x32_bf16(false, a1.v, false, b1.v, (short)0, acc11, false, false);
  }
  if (K2 > 0) {
    const unsigned short* arowB0 = A2 + dir * sA2 + (long)m0 * K2;
    const unsigned short* arowB1 = A2 + dir * sA2 + (long)m1 * K2;
    int nK2 = K2 >> 5;
    const unsigned short* w2a = W2 + dir * sW2 + ((long)nt * nK2) * 512 + lane * 16;
    const unsigned short* w2b = w2a + (long)nK2 * 512;
    for (int kt = 0; kt < nK2; ++kt) {
      Frag a0, a1, b0, b1;
      const unsigned short* ap0 = arowB0 + kt * 32 + 16 * hi;
      const unsigned short* ap1 = arowB1 + kt * 32 + 16 * hi;
      a0.q[0] = *(const uint4*)(ap0);
      a0.q[1] = *(const uint4*)(ap0 + 8);
      a1.q[0] = *(const uint4*)(ap1);
      a1.q[1] = *(const uint4*)(ap1 + 8);
      b0.q[0] = *(const uint4*)(w2a);
      b0.q[1] = *(const uint4*)(w2a + 8);
      b1.q[0] = *(const uint4*)(w2b);
      b1.q[1] = *(const uint4*)(w2b + 8);
      w2a += 512; w2b += 512;
      acc00 = __builtin_amdgcn_wmma_f32_16x16x32_bf16(false, a0.v, false, b0.v, (short)0, acc00, false, false);
      acc01 = __builtin_amdgcn_wmma_f32_16x16x32_bf16(false, a0.v, false, b1.v, (short)0, acc01, false, false);
      acc10 = __builtin_amdgcn_wmma_f32_16x16x32_bf16(false, a1.v, false, b0.v, (short)0, acc10, false, false);
      acc11 = __builtin_amdgcn_wmma_f32_16x16x32_bf16(false, a1.v, false, b1.v, (short)0, acc11, false, false);
    }
  }

  int n0 = nt * 16 + lrow;
  int n1 = n0 + 16;
  float bv0 = bias ? bias[dir * sBias + n0] : 0.0f;
  float bv1 = bias ? bias[dir * sBias + n1] : 0.0f;
#pragma unroll
  for (int r = 0; r < 8; ++r) {
    int mo0 = mt * 16 + r + 8 * hi;
    int mo1 = mo0 + 16;
    float v00 = acc00[r] + bv0;
    float v01 = acc01[r] + bv1;
    float v10 = acc10[r] + bv0;
    float v11 = acc11[r] + bv1;
    if (act == 1) { v00 = tanhf(v00); v01 = tanhf(v01); v10 = tanhf(v10); v11 = tanhf(v11); }
    if (Cf) {
      float* cf = Cf + dir * sCf;
      cf[(long)mo0 * ldc + n0] = v00;
      cf[(long)mo0 * ldc + n1] = v01;
      cf[(long)mo1 * ldc + n0] = v10;
      cf[(long)mo1 * ldc + n1] = v11;
    }
    if (Cb) {
      Cb[(long)mo0 * ldcb + n0] = f2bf(v00);
      Cb[(long)mo0 * ldcb + n1] = f2bf(v01);
      Cb[(long)mo1 * ldcb + n0] = f2bf(v10);
      Cb[(long)mo1 * ldcb + n1] = f2bf(v11);
    }
  }
}

// ---------------------------------------------------------------------------
// Elementwise LSTM gate update (i,f,g,o PyTorch order), masked for encoder.
// blockIdx.y = direction. enc=1: packed-sequence masking + write out sequence
// (fwd at pos t, bwd at pos len-1-t, only when t < len).
// ---------------------------------------------------------------------------
__global__ void lstm_update(
    const float* __restrict__ G, long sG,
    float* __restrict__ h, float* __restrict__ c, unsigned short* __restrict__ hb, long sS,
    const int* __restrict__ lens, int t, int enc,
    unsigned short* __restrict__ outSeq)
{
  int dir = blockIdx.y;
  int idx = blockIdx.x * blockDim.x + threadIdx.x;
  if (idx >= BB * HH) return;
  int b = idx >> 10, n = idx & (HH - 1);
  int len = 0;
  if (enc) { len = lens[b]; if (t >= len) return; }
  const float* g = G + dir * sG + (long)b * G4H;
  float gi = g[n], gf = g[HH + n], gg = g[2 * HH + n], go = g[3 * HH + n];
  long so = dir * sS + idx;
  float cn = sigm(gf) * c[so] + sigm(gi) * tanhf(gg);
  float hn = sigm(go) * tanhf(cn);
  c[so] = cn; h[so] = hn; hb[so] = f2bf(hn);
  if (enc) {
    int pos = dir ? (len - 1 - t) : t;
    outSeq[((long)pos * BB + b) * (2 * HH) + dir * HH + n] = f2bf(hn);
  }
}

__global__ void embed_x(const int* __restrict__ seqs, const float* __restrict__ embW,
                        unsigned short* __restrict__ xb) {
  long i = (long)blockIdx.x * blockDim.x + threadIdx.x;
  if (i >= (long)SS * BB * EE) return;
  int e = (int)(i % EE);
  long r = i / EE;
  int b = (int)(r % BB), s = (int)(r / BB);
  int tok = seqs[b * SS + s];
  xb[i] = f2bf(embW[(long)tok * EE + e]);
}

__global__ void init_states(const float* __restrict__ h0, const float* __restrict__ c0,
                            float* __restrict__ h, float* __restrict__ c,
                            unsigned short* __restrict__ hb) {
  int i = blockIdx.x * blockDim.x + threadIdx.x;
  if (i >= 4 * BB * HH) return;
  int n = i & (HH - 1);
  int k = i / (BB * HH);
  float hv = h0[k * HH + n], cv = c0[k * HH + n];
  h[i] = hv; c[i] = cv; hb[i] = f2bf(hv);
}

// hcat[l,b,:] = concat(state[l], state[l+2])  (states order l0f,l0b,l1f,l1b)
__global__ void build_cat_states(const float* __restrict__ st, unsigned short* __restrict__ dst) {
  int i = blockIdx.x * blockDim.x + threadIdx.x;
  if (i >= 2 * BB * 2 * HH) return;
  int j = i % (2 * HH);
  int r = i / (2 * HH);
  int b = r % BB, l = r / BB;
  int sidx = (j < HH) ? l : (l + 2);
  dst[i] = f2bf(st[(long)sidx * BB * HH + (long)b * HH + (j & (HH - 1))]);
}

__global__ void seq_mean_k(const float* __restrict__ attn, const int* __restrict__ lens,
                           float* __restrict__ sm) {
  int i = blockIdx.x * blockDim.x + threadIdx.x;
  if (i >= BB * HH) return;
  int b = i >> 10, n = i & (HH - 1);
  float s = 0.f;
  for (int t = 0; t < SS; ++t) s += attn[((long)t * BB + b) * HH + n];
  sm[i] = s / (float)lens[b];
}

__global__ void cat_f32(const float* __restrict__ a, const float* __restrict__ b2,
                        unsigned short* __restrict__ dst) {
  int i = blockIdx.x * blockDim.x + threadIdx.x;
  if (i >= BB * 2 * HH) return;
  int j = i % (2 * HH);
  int b = i / (2 * HH);
  float v = (j < HH) ? a[b * HH + j] : b2[b * HH + j - HH];
  dst[i] = f2bf(v);
}

__global__ void copy_h_to_cat(const unsigned short* __restrict__ src,
                              unsigned short* __restrict__ dst) {
  int i = blockIdx.x * blockDim.x + threadIdx.x;
  if (i >= BB * HH) return;
  int b = i >> 10, n = i & (HH - 1);
  dst[b * 2 * HH + n] = src[i];
}

__global__ void build_xin(const int* __restrict__ outSeqs, const float* __restrict__ dembW,
                          const unsigned short* __restrict__ prev,
                          unsigned short* __restrict__ xin, int t) {
  int i = blockIdx.x * blockDim.x + threadIdx.x;
  if (i >= BB * 2 * EE) return;
  int j = i % (2 * EE);
  int b = i / (2 * EE);
  unsigned short v;
  if (j < EE) { int tok = outSeqs[b * TT + t]; v = f2bf(dembW[(long)tok * EE + j]); }
  else v = prev[b * EE + (j - EE)];
  xin[i] = v;
}

__global__ void scores_k(const float* __restrict__ q, const float* __restrict__ attn,
                         const int* __restrict__ lens, float* __restrict__ sc) {
  __shared__ float red[128];
  int bs = blockIdx.x;
  int b = bs >> 7, s = bs & 127;
  int tid = threadIdx.x;
  float p = 0.f;
  const float* qr = q + (long)b * HH;
  const float* ar = attn + ((long)s * BB + b) * HH;
  for (int k = tid; k < HH; k += 128) p += qr[k] * ar[k];
  red[tid] = p; __syncthreads();
  for (int o = 64; o > 0; o >>= 1) { if (tid < o) red[tid] += red[tid + o]; __syncthreads(); }
  if (tid == 0) sc[bs] = (s < lens[b]) ? red[0] : -1e9f;
}

__global__ void softmax_k(const float* __restrict__ sc, float* __restrict__ w) {
  __shared__ float red[128];
  int b = blockIdx.x, s = threadIdx.x;
  float v = sc[b * SS + s];
  red[s] = v; __syncthreads();
  for (int o = 64; o > 0; o >>= 1) { if (s < o) red[s] = fmaxf(red[s], red[s + o]); __syncthreads(); }
  float mx = red[0]; __syncthreads();
  float e = expf(v - mx);
  red[s] = e; __syncthreads();
  for (int o = 64; o > 0; o >>= 1) { if (s < o) red[s] += red[s + o]; __syncthreads(); }
  w[b * SS + s] = e / red[0];
}

__global__ void ctx_k(const float* __restrict__ w, const float* __restrict__ attn,
                      unsigned short* __restrict__ cat2) {
  int b = blockIdx.x;
  int hcol = blockIdx.y * 256 + threadIdx.x;
  float acc = 0.f;
  for (int s = 0; s < SS; ++s) acc += w[b * SS + s] * attn[((long)s * BB + b) * HH + hcol];
  cat2[b * 2 * HH + HH + hcol] = f2bf(acc);
}

// ---------------------------------------------------------------------------
extern "C" void kernel_launch(void* const* d_in, const int* in_sizes, int n_in,
                              void* d_out, int out_size, void* d_ws, size_t ws_size,
                              hipStream_t stream) {
  const int*   input_seqs  = (const int*)d_in[0];
  const int*   input_lens  = (const int*)d_in[1];
  const int*   output_seqs = (const int*)d_in[2];
  const float* emb_W     = (const float*)d_in[3];
  const float* dec_emb_W = (const float*)d_in[4];
  const float* enc_Wih0  = (const float*)d_in[5];
  const float* enc_Whh0  = (const float*)d_in[6];
  const float* enc_b0    = (const float*)d_in[7];
  const float* enc_Wih1  = (const float*)d_in[8];
  const float* enc_Whh1  = (const float*)d_in[9];
  const float* enc_b1    = (const float*)d_in[10];
  const float* init_h0   = (const float*)d_in[11];
  const float* init_c0   = (const float*)d_in[12];
  const float* dec_Wih0  = (const float*)d_in[13];
  const float* dec_Whh0  = (const float*)d_in[14];
  const float* dec_b0    = (const float*)d_in[15];
  const float* dec_Wih1  = (const float*)d_in[16];
  const float* dec_Whh1  = (const float*)d_in[17];
  const float* dec_b1    = (const float*)d_in[18];
  const float* W_dec_h   = (const float*)d_in[19];
  const float* W_dec_c   = (const float*)d_in[20];
  const float* attn_wm   = (const float*)d_in[21];
  const float* attn_wq   = (const float*)d_in[22];
  const float* proj_W1   = (const float*)d_in[23];
  const float* proj_b1   = (const float*)d_in[24];
  const float* proj_W2   = (const float*)d_in[25];
  float* out = (float*)d_out;
  (void)in_sizes; (void)n_in; (void)out_size;

  char* wsb = (char*)d_ws;
  size_t off = 0;
  auto alloc = [&](size_t bytes) -> char* {
    off = (off + 255) & ~(size_t)255;
    char* p = wsb + off;
    off += bytes;
    return p;
  };

  // packed bf16 weights (~116 MB total -> fits MI455X 192MB L2)
  unsigned short* P_eWih0 = (unsigned short*)alloc(2L * 256 * 4096 * 2);
  unsigned short* P_eWhh0 = (unsigned short*)alloc(2L * 1024 * 4096 * 2);
  unsigned short* P_eWih1 = (unsigned short*)alloc(2L * 2048 * 4096 * 2);
  unsigned short* P_eWhh1 = (unsigned short*)alloc(2L * 1024 * 4096 * 2);
  unsigned short* P_dWih0 = (unsigned short*)alloc(512L * 4096 * 2);
  unsigned short* P_dWhh0 = (unsigned short*)alloc(1024L * 4096 * 2);
  unsigned short* P_dWih1 = (unsigned short*)alloc(1024L * 4096 * 2);
  unsigned short* P_dWhh1 = (unsigned short*)alloc(1024L * 4096 * 2);
  unsigned short* P_Wdh   = (unsigned short*)alloc(2048L * 1024 * 2);
  unsigned short* P_Wdc   = (unsigned short*)alloc(2048L * 1024 * 2);
  unsigned short* P_awm   = (unsigned short*)alloc(2048L * 1024 * 2);
  unsigned short* P_awq   = (unsigned short*)alloc(1024L * 1024 * 2);
  unsigned short* P_p1    = (unsigned short*)alloc(2048L * 1024 * 2);
  unsigned short* P_p2    = (unsigned short*)alloc(1024L * 256 * 2);
  unsigned short* P_demb  = (unsigned short*)alloc(256L * 2048 * 2);
  // activations / state
  unsigned short* x_bf  = (unsigned short*)alloc((long)SS * BB * EE * 2);
  unsigned short* out0  = (unsigned short*)alloc((long)SS * BB * 2048 * 2);
  unsigned short* out1  = (unsigned short*)alloc((long)SS * BB * 2048 * 2);
  float* attn   = (float*)alloc((long)SS * BB * HH * 4);
  float* enc_h  = (float*)alloc(4L * BB * HH * 4);
  float* enc_c  = (float*)alloc(4L * BB * HH * 4);
  unsigned short* enc_hb = (unsigned short*)alloc(4L * BB * HH * 2);
  float* G      = (float*)alloc(2L * BB * G4H * 4);
  float* sc     = (float*)alloc((long)BB * SS * 4);
  float* wsm    = (float*)alloc((long)BB * SS * 4);
  float* smean  = (float*)alloc((long)BB * HH * 4);
  unsigned short* hcat = (unsigned short*)alloc(2L * BB * 2048 * 2);
  unsigned short* ccat = (unsigned short*)alloc(2L * BB * 2048 * 2);
  float* dec_h  = (float*)alloc(2L * BB * HH * 4);
  float* dec_c  = (float*)alloc(2L * BB * HH * 4);
  unsigned short* dec_hb = (unsigned short*)alloc(2L * BB * HH * 2);
  unsigned short* cat2   = (unsigned short*)alloc((long)BB * 2048 * 2);
  unsigned short* xin    = (unsigned short*)alloc((long)BB * 512 * 2);
  unsigned short* p_bf   = (unsigned short*)alloc((long)BB * 1024 * 2);
  unsigned short* prev_bf = (unsigned short*)alloc((long)BB * 256 * 2);
  float* qbuf   = (float*)alloc((long)BB * HH * 4);

  if (off > ws_size) return;  // workspace too small; leave output poisoned

  auto pack = [&](const float* src, unsigned short* dst, int K, int N, int tr) {
    long total = (long)K * N;
    pack_w<<<dim3((unsigned)((total + 255) / 256)), dim3(256), 0, stream>>>(src, dst, K, N, tr);
  };
  auto gemm = [&](const unsigned short* A1, int K1, const unsigned short* W1,
                  const unsigned short* A2, int K2, const unsigned short* W2,
                  const float* bias, float* Cf, int ldc, unsigned short* Cb, int ldcb,
                  int M, int N, int act,
                  long sW1, long sW2, long sA2, int sBias, long sCf,
                  const int* lens, int t, int timeMode, int ndirs) {
    int tiles = (M / 32) * (N / 32);   // 2x2 macro-tiles, one per wave
    wmma_gemm<<<dim3((unsigned)((tiles + 3) / 4), (unsigned)ndirs), dim3(128), 0, stream>>>(
        A1, K1, W1, A2, K2, W2, bias, Cf, ldc, Cb, ldcb, M, N, act,
        sW1, sW2, sA2, sBias, sCf, lens, t, timeMode);
  };

  // ---- zero packed-sequence outputs (masked stores only fill valid slots) --
  hipMemsetAsync(out0, 0, (size_t)SS * BB * 2048 * 2, stream);
  hipMemsetAsync(out1, 0, (size_t)SS * BB * 2048 * 2, stream);

  // ---- pack all weights to bf16 fragment layout -----------------------------
  for (int d = 0; d < 2; ++d) {
    pack(enc_Wih0 + (long)d * 4096 * 256,  P_eWih0 + (long)d * 256 * 4096,  256,  4096, 1);
    pack(enc_Whh0 + (long)d * 4096 * 1024, P_eWhh0 + (long)d * 1024 * 4096, 1024, 4096, 1);
    pack(enc_Wih1 + (long)d * 4096 * 2048, P_eWih1 + (long)d * 2048 * 4096, 2048, 4096, 1);
    pack(enc_Whh1 + (long)d * 4096 * 1024, P_eWhh1 + (long)d * 1024 * 4096, 1024, 4096, 1);
  }
  pack(dec_Wih0, P_dWih0, 512, 4096, 1);
  pack(dec_Whh0, P_dWhh0, 1024, 4096, 1);
  pack(dec_Wih1, P_dWih1, 1024, 4096, 1);
  pack(dec_Whh1, P_dWhh1, 1024, 4096, 1);
  pack(W_dec_h, P_Wdh, 2048, 1024, 1);
  pack(W_dec_c, P_Wdc, 2048, 1024, 1);
  pack(attn_wm, P_awm, 2048, 1024, 0);
  pack(attn_wq, P_awq, 1024, 1024, 0);
  pack(proj_W1, P_p1, 2048, 1024, 1);
  pack(proj_W2, P_p2, 1024, 256, 1);
  pack(dec_emb_W, P_demb, 256, 2048, 1);

  // ---- embeddings + initial states -----------------------------------------
  embed_x<<<dim3((unsigned)(((long)SS * BB * EE + 255) / 256)), dim3(256), 0, stream>>>(
      input_seqs, emb_W, x_bf);
  init_states<<<dim3((4 * BB * HH + 255) / 256), dim3(256), 0, stream>>>(
      init_h0, init_c0, enc_h, enc_c, enc_hb);

  const long sDirW_ih0 = 256L * 4096, sDirW_hh = 1024L * 4096, sDirW_ih1 = 2048L * 4096;
  const long sDirState = (long)BB * HH, sDirG = (long)BB * G4H;

  // ---- encoder layer 0 (both directions fused via grid.y) ------------------
  for (int t = 0; t < SS; ++t) {
    gemm(x_bf, 256, P_eWih0, enc_hb, 1024, P_eWhh0, enc_b0,
         G, G4H, nullptr, 0, BB, G4H, 0,
         sDirW_ih0, sDirW_hh, sDirState, G4H, sDirG, input_lens, t, 1, 2);
    lstm_update<<<dim3(256, 2), dim3(256), 0, stream>>>(
        G, sDirG, enc_h, enc_c, enc_hb, sDirState, input_lens, t, 1, out0);
  }
  // ---- encoder layer 1 -----------------------------------------------------
  for (int t = 0; t < SS; ++t) {
    gemm(out0, 2048, P_eWih1, enc_hb + 2L * BB * HH, 1024, P_eWhh1, enc_b1,
         G, G4H, nullptr, 0, BB, G4H, 0,
         sDirW_ih1, sDirW_hh, sDirState, G4H, sDirG, input_lens, t, 1, 2);
    lstm_update<<<dim3(256, 2), dim3(256), 0, stream>>>(
        G, sDirG, enc_h + 2L * BB * HH, enc_c + 2L * BB * HH, enc_hb + 2L * BB * HH,
        sDirState, input_lens, t, 1, out1);
  }

  // ---- attention precompute: attn[s,b,:] = enc_out[s,b,:] @ attn_wm --------
  gemm(out1, 2048, P_awm, nullptr, 0, nullptr, nullptr,
       attn, HH, nullptr, 0, SS * BB, HH, 0, 0, 0, 0, 0, 0, nullptr, 0, 0, 1);
  seq_mean_k<<<dim3((BB * HH + 255) / 256), dim3(256), 0, stream>>>(attn, input_lens, smean);

  // ---- decoder init states -------------------------------------------------
  build_cat_states<<<dim3((2 * BB * 2048 + 255) / 256), dim3(256), 0, stream>>>(enc_h, hcat);
  build_cat_states<<<dim3((2 * BB * 2048 + 255) / 256), dim3(256), 0, stream>>>(enc_c, ccat);
  gemm(hcat, 2048, P_Wdh, nullptr, 0, nullptr, nullptr,
       dec_h, HH, dec_hb, HH, 2 * BB, HH, 0, 0, 0, 0, 0, 0, nullptr, 0, 0, 1);
  gemm(ccat, 2048, P_Wdc, nullptr, 0, nullptr, nullptr,
       dec_c, HH, nullptr, 0, 2 * BB, HH, 0, 0, 0, 0, 0, 0, nullptr, 0, 0, 1);
  // init_out = project(concat(init_h[1], seq_mean))
  cat_f32<<<dim3((BB * 2 * HH + 255) / 256), dim3(256), 0, stream>>>(dec_h + (long)BB * HH, smean, cat2);
  gemm(cat2, 2048, P_p1, nullptr, 0, nullptr, proj_b1,
       nullptr, 0, p_bf, HH, BB, HH, 1, 0, 0, 0, 0, 0, nullptr, 0, 0, 1);
  gemm(p_bf, 1024, P_p2, nullptr, 0, nullptr, nullptr,
       nullptr, 0, prev_bf, EE, BB, EE, 0, 0, 0, 0, 0, 0, nullptr, 0, 0, 1);

  // ---- decoder loop --------------------------------------------------------
  for (int t = 0; t < TT; ++t) {
    build_xin<<<dim3((BB * 512 + 255) / 256), dim3(256), 0, stream>>>(
        output_seqs, dec_emb_W, prev_bf, xin, t);
    // cell 0
    gemm(xin, 512, P_dWih0, dec_hb, 1024, P_dWhh0, dec_b0,
         G, G4H, nullptr, 0, BB, G4H, 0, 0, 0, 0, 0, 0, nullptr, 0, 0, 1);
    lstm_update<<<dim3(256, 1), dim3(256), 0, stream>>>(
        G, 0, dec_h, dec_c, dec_hb, 0, nullptr, 0, 0, nullptr);
    // cell 1 (input = new h0)
    gemm(dec_hb, 1024, P_dWih1, dec_hb + (long)BB * HH, 1024, P_dWhh1, dec_b1,
         G, G4H, nullptr, 0, BB, G4H, 0, 0, 0, 0, 0, 0, nullptr, 0, 0, 1);
    lstm_update<<<dim3(256, 1), dim3(256), 0, stream>>>(
        G, 0, dec_h + (long)BB * HH, dec_c + (long)BB * HH, dec_hb + (long)BB * HH,
        0, nullptr, 0, 0, nullptr);
    // attention
    gemm(dec_hb + (long)BB * HH, 1024, P_awq, nullptr, 0, nullptr, nullptr,
         qbuf, HH, nullptr, 0, BB, HH, 0, 0, 0, 0, 0, 0, nullptr, 0, 0, 1);
    scores_k<<<dim3(BB * SS), dim3(128), 0, stream>>>(qbuf, attn, input_lens, sc);
    softmax_k<<<dim3(BB), dim3(128), 0, stream>>>(sc, wsm);
    ctx_k<<<dim3(BB, 4), dim3(256), 0, stream>>>(wsm, attn, cat2);
    copy_h_to_cat<<<dim3((BB * HH + 255) / 256), dim3(256), 0, stream>>>(
        dec_hb + (long)BB * HH, cat2);
    // projection + logits
    gemm(cat2, 2048, P_p1, nullptr, 0, nullptr, proj_b1,
         nullptr, 0, p_bf, HH, BB, HH, 1, 0, 0, 0, 0, 0, nullptr, 0, 0, 1);
    gemm(p_bf, 1024, P_p2, nullptr, 0, nullptr, nullptr,
         nullptr, 0, prev_bf, EE, BB, EE, 0, 0, 0, 0, 0, 0, nullptr, 0, 0, 1);
    gemm(prev_bf, 256, P_demb, nullptr, 0, nullptr, nullptr,
         out + (long)t * AAV, TT * AAV, nullptr, 0, BB, AAV, 0,
         0, 0, 0, 0, 0, nullptr, 0, 0, 1);
  }
}
